// SAM_42322607734846
// MI455X (gfx1250) — matrix operand order repaired and checked by reading, not compiled, not run on hardware
//
#include <hip/hip_runtime.h>

typedef _Float16 half_t;
typedef __attribute__((ext_vector_type(16))) _Float16 v16h;
typedef __attribute__((ext_vector_type(8)))  _Float16 v8h;
typedef __attribute__((ext_vector_type(8)))  float    v8f;
typedef __attribute__((ext_vector_type(4)))  unsigned int v4u;
typedef __attribute__((ext_vector_type(8)))  unsigned int v8u;

#define EPSV 1e-5f
constexpr int B_   = 8;
constexpr int CIN  = 256;
constexpr int H_   = 56;
constexpr int W_   = 56;
constexpr int NPIX = H_ * W_;          // 3136
constexpr int REL  = 32;
constexpr int OUT  = 256;
constexpr int WC   = 32;
constexpr int MID  = 320;              // rel*(9+1)
constexpr int WBM  = 288;              // 9*wc

// ---- workspace layout (bytes) ----
constexpr size_t OFF_Y   = 0;                                   // f32  B*320*N  (rows 0..31 x1, 32..63 x2raw, 64..319 x3)
constexpr size_t OFF_WW  = OFF_Y  + (size_t)B_ * MID * NPIX * 4;  // f32  B*288*N  (attention weights w)
constexpr size_t OFF_WC  = OFF_WW + (size_t)B_ * WBM * NPIX * 4;  // f16  320*256
constexpr size_t OFF_WA  = OFF_WC + (size_t)MID * CIN * 2;        // f16  32*320
constexpr size_t OFF_WB  = OFF_WA + (size_t)WC * MID * 2;         // f16  288*32
constexpr size_t OFF_BC  = OFF_WB + (size_t)WBM * WC * 2;         // f32  320 concat bias
constexpr size_t OFF_S1  = OFF_BC + 320 * 4;                      // f32  320 bn1 scale
constexpr size_t OFF_T1  = OFF_S1 + 320 * 4;                      // f32  320 bn1 shift
constexpr size_t OFF_S2  = OFF_T1 + 320 * 4;                      // f32  32  bn2 scale
constexpr size_t OFF_T2  = OFF_S2 + 32 * 4;                       // f32  32  bn2 shift

__device__ __forceinline__ int reflect_i(int v, int n) {
    v = v < 0 ? -v : v;
    v = v >= n ? 2 * n - 2 - v : v;
    return v;
}

// ------------------------------------------------------------------
// Kernel 0: fold weights to f16, precompute BN scale/shift
// ------------------------------------------------------------------
__global__ void prep_kernel(const float* __restrict__ w1, const float* __restrict__ w2,
                            const float* __restrict__ w3, const float* __restrict__ b1,
                            const float* __restrict__ b2, const float* __restrict__ b3,
                            const float* __restrict__ g1, const float* __restrict__ be1,
                            const float* __restrict__ m1, const float* __restrict__ v1,
                            const float* __restrict__ wA, const float* __restrict__ g2,
                            const float* __restrict__ be2, const float* __restrict__ m2,
                            const float* __restrict__ v2, const float* __restrict__ wB,
                            half_t* __restrict__ Wc, half_t* __restrict__ WAh,
                            half_t* __restrict__ WBh, float* __restrict__ biascat,
                            float* __restrict__ s1, float* __restrict__ t1,
                            float* __restrict__ s2, float* __restrict__ t2)
{
    int i = blockIdx.x * blockDim.x + threadIdx.x;
    if (i < MID * CIN) {
        int m = i / CIN, k = i % CIN;
        float v = (m < REL)     ? w1[m * CIN + k]
                : (m < 2 * REL) ? w2[(m - REL) * CIN + k]
                                : w3[(m - 2 * REL) * CIN + k];
        Wc[i] = (half_t)v;
    }
    if (i < WC * MID)  WAh[i] = (half_t)wA[i];
    if (i < WBM * WC)  WBh[i] = (half_t)wB[i];
    if (i < MID) {
        biascat[i] = (i < REL) ? b1[i] : (i < 2 * REL) ? b2[i - REL] : b3[i - 2 * REL];
        float sc = g1[i] * rsqrtf(v1[i] + EPSV);
        s1[i] = sc;
        t1[i] = be1[i] - m1[i] * sc;
    }
    if (i < WC) {
        float sc = g2[i] * rsqrtf(v2[i] + EPSV);
        s2[i] = sc;
        t2[i] = be2[i] - m2[i] * sc;
    }
}

// ------------------------------------------------------------------
// Kernel 1: Y = Wcat(320x256) @ X(256xN) + biascat  (per batch)
// The block's 256x64 f32 X tile (64 KB) is DMA'd into LDS once via the
// Tensor Data Mover; 8 waves (2 M-tiles x 4 N-tiles) consume it.
// ------------------------------------------------------------------
__global__ __launch_bounds__(256)
void qkv_gemm(const float* __restrict__ x, const half_t* __restrict__ Wc,
              const float* __restrict__ biascat, float* __restrict__ Y)
{
    __shared__ float xtile[CIN * 64];                 // 64 KB, [k][n_local]
    const int lane = threadIdx.x & 31;
    const int wave = threadIdx.x >> 5;
    const int b    = blockIdx.z;
    const int n0b  = blockIdx.x * 64;
    const int m0   = blockIdx.y * 32 + (wave >> 2) * 16;
    const int nt   = wave & 3;
    const int col  = lane & 15;
    const int hi16 = lane >> 4;   // 0: lanes 0-15, 1: lanes 16-31

    if (wave == 0) {
        // ---- Tensor DMA descriptor (D#), 2-D tile: 64 cols x 256 rows, f32 ----
        unsigned long long gaddr =
            (unsigned long long)(uintptr_t)(x + (size_t)b * CIN * NPIX + n0b);
        unsigned int laddr = (unsigned int)(uintptr_t)(void*)xtile; // LDS offset in addr[31:0]
        v4u g0;
        g0[0] = 1u;                                           // count=1, user descriptor
        g0[1] = laddr;                                        // lds_addr
        g0[2] = (unsigned int)gaddr;                          // global_addr[31:0]
        g0[3] = (unsigned int)((gaddr >> 32) & 0x1FFFFFFu)    // global_addr[56:32]
              | (2u << 30);                                   // type = 2 ("image")
        v8u g1;
        g1[0] = 2u << 16;                 // data_size = 2 (4 bytes); no pad/iter/multicast
        g1[1] = (unsigned int)NPIX << 16; // tensor_dim0[15:0]   (3136)
        g1[2] = (unsigned int)CIN << 16;  // tensor_dim0[31:16]=0 | tensor_dim1[15:0]=256
        g1[3] = 64u << 16;                // tensor_dim1[31:16]=0 | tile_dim0=64
        g1[4] = (unsigned int)CIN;        // tile_dim1=256 | tile_dim2=0
        g1[5] = (unsigned int)NPIX;       // tensor_dim0_stride[31:0] (elements)
        g1[6] = 0u;                       // stride hi | tensor_dim1_stride lo (unused, 2-D)
        g1[7] = 0u;
        asm volatile("tensor_load_to_lds %0, %1" :: "s"(g0), "s"(g1) : "memory");
        __builtin_amdgcn_s_wait_tensorcnt(0);
    }
    __syncthreads();

    v8f acc = {};
    for (int kk = 0; kk < CIN; kk += 32) {
        // A fragment (16x32 f16): lane<16 -> K 0..15, lane>=16 -> K 16..31
        const half_t* arow = Wc + (size_t)(m0 + col) * CIN + kk + hi16 * 16;
        v16h a;
        *(v8h*)&a       = *(const v8h*)(arow);
        *((v8h*)&a + 1) = *(const v8h*)(arow + 8);
        // B fragment (32x16 f16) from the LDS tile
        const float* xc = &xtile[(kk + hi16 * 16) * 64 + nt * 16 + col];
        v16h bf;
#pragma unroll
        for (int j = 0; j < 16; ++j)
            bf[j] = (half_t)xc[j * 64];
        acc = __builtin_amdgcn_wmma_f32_16x16x32_f16(false, a, false, bf,
                                                     (short)0, acc, false, false);
    }
    float* Yb = Y + (size_t)b * MID * NPIX;
#pragma unroll
    for (int r = 0; r < 8; ++r) {
        int m = m0 + r + hi16 * 8;
        Yb[(size_t)m * NPIX + n0b + nt * 16 + col] = acc[r] + biascat[m];
    }
}

// ------------------------------------------------------------------
// Kernel 2: fused  feat-gather -> BN1+ReLU -> wA GEMM -> BN2+ReLU
//                  -> repack -> wB GEMM + bB  -> Wout (B,288,N)
// One wave handles 16 pixels; 8 waves/block = 128 pixels.
// ------------------------------------------------------------------
__global__ __launch_bounds__(256)
void attn_weights(const float* __restrict__ Y, const half_t* __restrict__ WAh,
                  const half_t* __restrict__ WBh,
                  const float* __restrict__ s1, const float* __restrict__ t1,
                  const float* __restrict__ s2, const float* __restrict__ t2,
                  const float* __restrict__ bB, float* __restrict__ Wout)
{
    const int lane = threadIdx.x & 31;
    const int wave = threadIdx.x >> 5;
    const int b    = blockIdx.y;
    const int n0   = blockIdx.x * 128 + wave * 16;
    if (n0 >= NPIX) return;                       // uniform per wave
    const int col  = lane & 15;
    const int hi16 = lane >> 4;
    const int n    = n0 + col;
    const int hi   = n / W_, wi = n % W_;
    const float* Yb = Y + (size_t)b * MID * NPIX;

    int noff[9];
#pragma unroll
    for (int t = 0; t < 9; ++t) {
        int h2 = reflect_i(hi + t / 3 - 1, H_);
        int w2 = reflect_i(wi + t % 3 - 1, W_);
        noff[t] = h2 * W_ + w2;
    }

    // ---- h = wA @ relu(bn1(feat)), K = 320 in 10 steps of 32 ----
    v8f acc0 = {}, acc1 = {};
    for (int s = 0; s < MID; s += 32) {
        v16h bf;
#pragma unroll
        for (int j = 0; j < 16; ++j) {
            int r = s + hi16 * 16 + j;          // feat row 0..319
            float v;
            if (r < REL) {
                v = Yb[(size_t)r * NPIX + n];   // x1 channel
            } else {
                int u = r - REL;                // unfold: 32 + c*9 + t
                int c = u / 9, t = u % 9;
                v = Yb[(size_t)(REL + c) * NPIX + noff[t]];
            }
            v = v * s1[r] + t1[r];
            v = v > 0.f ? v : 0.f;
            bf[j] = (half_t)v;
        }
        const half_t* ar0 = WAh + (size_t)col * MID + s + hi16 * 16;
        const half_t* ar1 = WAh + (size_t)(16 + col) * MID + s + hi16 * 16;
        v16h a0, a1;
        *(v8h*)&a0 = *(const v8h*)ar0;  *((v8h*)&a0 + 1) = *(const v8h*)(ar0 + 8);
        *(v8h*)&a1 = *(const v8h*)ar1;  *((v8h*)&a1 + 1) = *(const v8h*)(ar1 + 8);
        acc0 = __builtin_amdgcn_wmma_f32_16x16x32_f16(false, a0, false, bf,
                                                      (short)0, acc0, false, false);
        acc1 = __builtin_amdgcn_wmma_f32_16x16x32_f16(false, a1, false, bf,
                                                      (short)0, acc1, false, false);
    }

    // ---- bn2 + relu (elementwise on C-layout accumulators) ----
#pragma unroll
    for (int r = 0; r < 8; ++r) {
        int m = r + hi16 * 8;
        float v0 = acc0[r] * s2[m] + t2[m];
        float v1 = acc1[r] * s2[16 + m] + t2[16 + m];
        acc0[r] = v0 > 0.f ? v0 : 0.f;
        acc1[r] = v1 > 0.f ? v1 : 0.f;
    }

    // ---- repack h (32x16, C layout) into next B fragment via xor-16 ----
    // B here takes rows 0..15 in lanes 0-15, rows 16..31 in lanes 16-31 (same k order as A), same column.
    v16h hb;
#pragma unroll
    for (int j = 0; j < 8; ++j) {
        float a0p = __shfl_xor(acc0[j], 16);
        float a1p = __shfl_xor(acc1[j], 16);
        hb[j]     = (half_t)(hi16 ? a1p     : acc0[j]);
        hb[j + 8] = (half_t)(hi16 ? acc1[j] : a0p);
    }

    // ---- w = wB(288x32) @ h + bB : 18 M-tiles, single K step ----
    float* Wb = Wout + (size_t)b * WBM * NPIX;
    for (int tmt = 0; tmt < 18; ++tmt) {
        const half_t* ar = WBh + (size_t)(tmt * 16 + col) * WC + hi16 * 16;
        v16h a;
        *(v8h*)&a = *(const v8h*)ar;  *((v8h*)&a + 1) = *(const v8h*)(ar + 8);
        v8f c = {};
        c = __builtin_amdgcn_wmma_f32_16x16x32_f16(false, a, false, hb,
                                                   (short)0, c, false, false);
#pragma unroll
        for (int r = 0; r < 8; ++r) {
            int m = tmt * 16 + r + hi16 * 8;
            Wb[(size_t)m * NPIX + n] = c[r] + bB[m];
        }
    }
}

// ------------------------------------------------------------------
// Kernel 3: y[b, g*32+c, n] = sum_t w[b, c*9+t, n] * x3[b, g*32+c, n'(t)]
// x3 lives in Y rows 64..319. Everything is L2-resident (192 MB).
// ------------------------------------------------------------------
__global__ __launch_bounds__(256)
void aggregate(const float* __restrict__ Y, const float* __restrict__ Wout,
               float* __restrict__ out)
{
    size_t idx = (size_t)blockIdx.x * blockDim.x + threadIdx.x;
    int n = (int)(idx % NPIX);
    size_t r = idx / NPIX;
    int o = (int)(r % OUT);
    int b = (int)(r / OUT);
    if (b >= B_) return;
    int c  = o & 31;
    int hi = n / W_, wi = n % W_;
    const float* x3 = Y + (size_t)b * MID * NPIX + (size_t)(2 * REL + o) * NPIX;
    const float* wr = Wout + (size_t)b * WBM * NPIX + (size_t)(c * 9) * NPIX + n;
    float acc = 0.f;
#pragma unroll
    for (int t = 0; t < 9; ++t) {
        int h2 = reflect_i(hi + t / 3 - 1, H_);
        int w2 = reflect_i(wi + t % 3 - 1, W_);
        acc += wr[(size_t)t * NPIX] * x3[h2 * W_ + w2];
    }
    out[idx] = acc;
}

// ------------------------------------------------------------------
extern "C" void kernel_launch(void* const* d_in, const int* in_sizes, int n_in,
                              void* d_out, int out_size, void* d_ws, size_t ws_size,
                              hipStream_t stream)
{
    const float* x   = (const float*)d_in[0];
    const float* w1  = (const float*)d_in[1];
    const float* b1  = (const float*)d_in[2];
    const float* w2  = (const float*)d_in[3];
    const float* b2  = (const float*)d_in[4];
    const float* w3  = (const float*)d_in[5];
    const float* b3  = (const float*)d_in[6];
    const float* g1  = (const float*)d_in[7];
    const float* be1 = (const float*)d_in[8];
    const float* m1  = (const float*)d_in[9];
    const float* v1  = (const float*)d_in[10];
    const float* wA  = (const float*)d_in[11];
    const float* g2  = (const float*)d_in[12];
    const float* be2 = (const float*)d_in[13];
    const float* m2  = (const float*)d_in[14];
    const float* v2  = (const float*)d_in[15];
    const float* wB  = (const float*)d_in[16];
    const float* bB  = (const float*)d_in[17];

    char* ws = (char*)d_ws;
    float*  Y       = (float*)(ws + OFF_Y);
    float*  Wout    = (float*)(ws + OFF_WW);
    half_t* Wc      = (half_t*)(ws + OFF_WC);
    half_t* WAh     = (half_t*)(ws + OFF_WA);
    half_t* WBh     = (half_t*)(ws + OFF_WB);
    float*  biascat = (float*)(ws + OFF_BC);
    float*  s1      = (float*)(ws + OFF_S1);
    float*  t1      = (float*)(ws + OFF_T1);
    float*  s2      = (float*)(ws + OFF_S2);
    float*  t2      = (float*)(ws + OFF_T2);

    prep_kernel<<<(MID * CIN + 255) / 256, 256, 0, stream>>>(
        w1, w2, w3, b1, b2, b3, g1, be1, m1, v1, wA, g2, be2, m2, v2, wB,
        Wc, WAh, WBh, biascat, s1, t1, s2, t2);

    dim3 grid1(NPIX / 64, MID / 32, B_);            // 49 x 10 x 8
    qkv_gemm<<<grid1, 256, 0, stream>>>(x, Wc, biascat, Y);

    dim3 grid2((NPIX + 127) / 128, B_);             // 25 x 8
    attn_weights<<<grid2, 256, 0, stream>>>(Y, WAh, WBh, s1, t1, s2, t2, bB, Wout);

    size_t total = (size_t)B_ * OUT * NPIX;         // 6,422,528
    aggregate<<<(unsigned)((total + 255) / 256), 256, 0, stream>>>(Y, Wout, (float*)d_out);
}
